// StateSpaceDiffusionModel_52003464020565
// MI455X (gfx1250) — matrix operand order repaired and checked by reading, not compiled, not run on hardware
//
#include <hip/hip_runtime.h>
#include <hip/hip_bf16.h>

// ---------------------------------------------------------------------------
// StateSpaceDiffusion: y[b,h,t] = sum_{s<=t} u[b,h,s] * f[h,t-s]
// where f[h,t] = k_h . (M_h^t e0) and M_h is the L1-normalized companion
// matrix (sparse: col j has knorm_j/s_j at row 0 and 1/s_j at row j+1).
//
// Kernel 1: per-head sparse Krylov recurrence (wave32, shfl-based), emits
//           the filter reversed in f16 to workspace.
// Kernel 2: per-head causal conv as batch-GEMM vs Toeplitz(f) using
//           v_wmma_f32_16x16x32_f16 (M=16 batches, N=16 timesteps, K=32),
//           scalar (SGPR) K-loop + double-buffered LDS operand fetch.
// ---------------------------------------------------------------------------

typedef _Float16 h4  __attribute__((ext_vector_type(4)));
typedef _Float16 h8  __attribute__((ext_vector_type(8)));
typedef _Float16 v16h __attribute__((ext_vector_type(16)));
typedef float    v8f  __attribute__((ext_vector_type(8)));

#define HEADS 512
#define NSTATE 64
#define SEQLEN 2048
#define NBATCH 16
#define SLAB 512
#define USTRIDE 520            // 512 + 8 pad: rows hit distinct LDS banks
#define FRLEN 2112             // 2048 + 16 front pad + 31 tail pad, rounded

// ---------------------------------------------------------------------------
// Kernel 1: filter generation. One wave32 per head; lane L owns state
// elements L and L+32.  x_{t+1}[0] = g.x ; x_{t+1}[i] = c_{i-1} x[i-1].
// Stores fr[h][j] = (f16) f[h][2047 - j]   (reversed for Toeplitz gathers).
// ---------------------------------------------------------------------------
__global__ __launch_bounds__(32, 4)
void ssd_filter_kernel(const float* __restrict__ k, _Float16* __restrict__ fr)
{
    const int h = blockIdx.x;
    const int lane = threadIdx.x;
    const float* kh = k + h * NSTATE;

    const float k0 = kh[lane];
    const float k1 = kh[lane + 32];

    // k_norm = clip(k, 1/16, 1) / sum(|clip|)
    float kc0 = fminf(fmaxf(k0, 0.0625f), 1.0f);
    float kc1 = fminf(fmaxf(k1, 0.0625f), 1.0f);
    float ssum = fabsf(kc0) + fabsf(kc1);
    #pragma unroll
    for (int o = 16; o; o >>= 1) ssum += __shfl_xor(ssum, o, 32);
    const float kn0 = kc0 / ssum;
    const float kn1 = kc1 / ssum;

    // column sums of A + b k^T : s_j = knorm_j + (j < 63)
    const float sc0 = kn0 + 1.0f;                          // j = lane  (< 63)
    const float sc1 = kn1 + ((lane == 31) ? 0.0f : 1.0f);  // j = lane+32
    const float c0 = 1.0f / sc0, c1 = 1.0f / sc1;          // subdiag scale
    const float g0 = kn0 * c0,   g1 = kn1 * c1;            // row-0 weights

    // shift multipliers: new_x[i] = c[i-1] * x[i-1]
    float cm0 = __shfl_up(c0, 1, 32);          // c[lane-1]      (lane>0)
    float cm1 = __shfl_up(c1, 1, 32);          // c[lane+31]     (lane>0)
    const float c0_31 = __shfl(c0, 31, 32);
    if (lane == 0) cm1 = c0_31;                // c[31] feeds element 32

    float x0 = (lane == 0) ? 1.0f : 0.0f;      // x = e0
    float x1 = 0.0f;
    float fbuf = 0.0f;

    _Float16* out = fr + (size_t)h * SEQLEN;

    for (int t = 0; t < SEQLEN; ++t) {
        float kx = k0 * x0 + k1 * x1;          // f_t = k . x   (original k)
        float gx = g0 * x0 + g1 * x1;          // next x[0]
        #pragma unroll
        for (int o = 16; o; o >>= 1) {
            kx += __shfl_xor(kx, o, 32);
            gx += __shfl_xor(gx, o, 32);
        }
        if (lane == (t & 31)) fbuf = kx;       // bank 32 results
        if ((t & 31) == 31)                    // coalesced reversed store
            out[2047 - (t - 31) - lane] = (_Float16)fbuf;

        // advance state: shift+scale, inject gx at element 0
        const float xu0 = __shfl_up(x0, 1, 32);
        const float xu1 = __shfl_up(x1, 1, 32);
        const float x0_31 = __shfl(x0, 31, 32);
        const float nx1 = cm1 * ((lane == 0) ? x0_31 : xu1);
        const float nx0 = (lane == 0) ? gx : cm0 * xu0;
        x0 = nx0;
        x1 = nx1;
    }
}

// ---------------------------------------------------------------------------
// Kernel 2: per-head causal conv as WMMA batch-GEMM.
//   Y'[b, t] = sum_s U'[b, s] * F'[s, t],  F'[s,t] = f[t-s]  (0 if t<s)
// A tile (16x32 f16)  = u rows (batch x K)  -> aligned ds_load_b128 x2
// B tile (32x16 f16)  = Toeplitz slice      -> aligned ds_load_2addr_b64 x2
//                       from phase-replicated reversed filter in LDS
// C/D  (16x16 f32)    = v8f accumulators, 16 tiles/wave held in VGPRs
// ---------------------------------------------------------------------------
__global__ __launch_bounds__(256, 1)
void ssd_conv_kernel(const float* __restrict__ u,
                     const _Float16* __restrict__ fr,
                     float* __restrict__ y)
{
    __shared__ __align__(16) _Float16 Us[NBATCH][USTRIDE];  // u slab, f16
    __shared__ __align__(16) _Float16 FRr[4][FRLEN];        // filter, 4 phases

    const int h    = blockIdx.x;
    const int tid  = threadIdx.x;
    const int lane = tid & 31;
    // wave id is uniform across the wave: pin it into an SGPR so that the
    // K-loop bounds below are scalar (s_cbranch loop, EXEC stays all-ones).
    const int wave = __builtin_amdgcn_readfirstlane(tid >> 5);   // 8 waves

    // --- build phase-replicated reversed filter:
    // FR(j) = f[2063-j] for 16<=j<2064 else 0 ;  FRr[r][i] = FR(i+r)
    {
        const _Float16* frh = fr + (size_t)h * SEQLEN;
        for (int rr = 0; rr < 4; ++rr) {
            for (int i = tid; i < FRLEN; i += 256) {
                const int j = i + rr;
                _Float16 v = (_Float16)0.0f;
                if (j >= 16 && j < 2064) v = frh[j - 16];   // = f[2063-j]
                FRr[rr][i] = v;
            }
        }
    }

    v8f acc[16];
    #pragma unroll
    for (int j = 0; j < 16; ++j)
        #pragma unroll
        for (int e = 0; e < 8; ++e) acc[j][e] = 0.0f;

    const int m     = lane & 15;               // A row / B col within tile
    const int koffA = (lane & 16) ? 16 : 0;    // A K-slot base per lane half
    const int kb    = (lane & 16) ? 16 : 0;    // B K base per lane half

    // operand fetch helpers (all-lane, no divergence)
    auto loadA = [&](int ls) -> v16h {
        const h8 alo = *(const h8*)&Us[m][ls + koffA];
        const h8 ahi = *(const h8*)&Us[m][ls + 8 + koffA];
        return __builtin_shufflevector(alo, ahi,
                                       0,1,2,3,4,5,6,7,8,9,10,11,12,13,14,15);
    };
    auto loadB = [&](int t0, int s0) -> v16h {
        // B[k][n] = f[t0+n-s0-k] = FR(2063 - t0 - n + s0 + k), k ascending
        const int j0   = 2063 - t0 - m + s0 + kb;
        const int rsel = j0 & 3;                 // phase copy -> 8B aligned
        const h4* bp = (const h4*)&FRr[rsel][j0 - rsel];
        const h8 blo = __builtin_shufflevector(bp[0], bp[1], 0,1,2,3,4,5,6,7);
        const h8 bhi = __builtin_shufflevector(bp[2], bp[3], 0,1,2,3,4,5,6,7);
        return __builtin_shufflevector(blo, bhi,
                                       0,1,2,3,4,5,6,7,8,9,10,11,12,13,14,15);
    };

    for (int slab = 0; slab < 4; ++slab) {
        const int sbase = slab * SLAB;

        // --- stage u slab -> LDS (f32 -> f16), 32 elems / thread
        {
            const int row = tid >> 4;                   // batch
            const int c0  = (tid & 15) * 32;
            const float* gp = u + ((size_t)row * HEADS + h) * SEQLEN + sbase + c0;
            const float4* up = (const float4*)gp;
            #pragma unroll
            for (int q = 0; q < 8; ++q) {
                const float4 v = up[q];
                h4 p = { (_Float16)v.x, (_Float16)v.y,
                         (_Float16)v.z, (_Float16)v.w };
                *(h4*)&Us[row][c0 + q * 4] = p;
            }
            if (slab < 3)   // global_prefetch_b8 of next slab
                __builtin_prefetch(gp + SLAB, 0, 1);
        }
        __syncthreads();

        // --- triangular K-loop over this slab for the wave's 16 tiles
        #pragma unroll
        for (int j = 0; j < 16; ++j) {
            const int t0 = j * 128 + wave * 16;         // scalar (SGPR)
            if (t0 + 16 <= sbase) continue;             // slab past s-range
            int send = t0 + 16;                         // s0 < t0+16 (causal)
            if (send > sbase + SLAB) send = sbase + SLAB;

            // double-buffered: issue next chunk's 4 DS loads before the
            // WMMA consuming the current chunk, so only dscnt<=4 is waited.
            int s0 = sbase;
            v16h Acur = loadA(s0 - sbase);
            v16h Bcur = loadB(t0, s0);
            for (s0 += 32; s0 < send; s0 += 32) {
                const v16h Anx = loadA(s0 - sbase);
                const v16h Bnx = loadB(t0, s0);
                acc[j] = __builtin_amdgcn_wmma_f32_16x16x32_f16(
                    false, Acur, false, Bcur, (short)0, acc[j], false, false);
                Acur = Anx;
                Bcur = Bnx;
            }
            acc[j] = __builtin_amdgcn_wmma_f32_16x16x32_f16(
                false, Acur, false, Bcur, (short)0, acc[j], false, false);
        }
        __syncthreads();
    }

    // --- store: C/D layout: VGPR r -> batch r (+8 for lanes 16-31), N = lane&15
    const int mh = (lane & 16) ? 8 : 0;
    #pragma unroll
    for (int j = 0; j < 16; ++j) {
        const int t0 = j * 128 + wave * 16;
        #pragma unroll
        for (int r8 = 0; r8 < 8; ++r8) {
            const int b = r8 + mh;
            y[((size_t)b * HEADS + h) * SEQLEN + t0 + (lane & 15)] = acc[j][r8];
        }
    }
}

// ---------------------------------------------------------------------------
extern "C" void kernel_launch(void* const* d_in, const int* in_sizes, int n_in,
                              void* d_out, int out_size, void* d_ws, size_t ws_size,
                              hipStream_t stream)
{
    const float* u = (const float*)d_in[0];   // (16, 512, 2048) f32
    const float* k = (const float*)d_in[1];   // (1, 512, 64)    f32
    float* y = (float*)d_out;                 // (16, 512, 2048) f32

    // workspace: reversed f16 filter, 512*2048*2 = 2 MiB
    _Float16* fr = (_Float16*)d_ws;

    ssd_filter_kernel<<<HEADS, 32, 0, stream>>>(k, fr);
    ssd_conv_kernel<<<HEADS, 256, 0, stream>>>(u, fr, y);
}
